// GCN_11665131176187
// MI455X (gfx1250) — matrix operand order repaired and checked by reading, not compiled, not run on hardware
//
#include <hip/hip_runtime.h>
#include <hip/hip_bf16.h>

// GCN forward for MI455X (gfx1250, wave32).
// Strategy: L2-resident pipeline (working set ~55MB << 192MB L2).
//   - Node transforms via v_wmma_f32_16x16x32_f16 (8 WMMAs / wave, 16x64 tile).
//   - W staged to LDS *pre-packed in B-fragment layout* -> fragment loads are
//     2x ds_load_b128 each (was 16x ds_load_u16 + b16 merges).
//   - Edge gather/scatter via global_atomic_add_f32 (memory-bound stage).

#define F_DIM 64
#define NGRAPH 256

typedef __attribute__((ext_vector_type(16))) _Float16 v16h;
typedef __attribute__((ext_vector_type(8)))  float    v8f;

// ---------------- degree / normalization ----------------

__global__ void deg_init(float* deg, int n) {
    int i = blockIdx.x * 256 + threadIdx.x;
    if (i < n) deg[i] = 1.0f;                      // self-loop contributes 1
}

__global__ void deg_accum(const long long* __restrict__ dst, float* deg, int E) {
    int e = blockIdx.x * 256 + threadIdx.x;
    if (e < E) atomicAdd(&deg[(int)dst[e]], 1.0f);
}

__global__ void dinv_kernel(const float* __restrict__ deg, float* dinv, int n) {
    int i = blockIdx.x * 256 + threadIdx.x;
    if (i < n) {
        float d = deg[i];
        dinv[i] = (d > 0.0f) ? rsqrtf(d) : 0.0f;
    }
}

// ---------------- WMMA GEMM: H[N x 64] = X[N x 64] * W[64 x 64] ----------------
// One wave computes a 16x64 output tile. N must be a multiple of 16 (50000 is).
// Layouts per CDNA5 ISA 7.12.2:
//   A (16x32 f16): lane l -> row m=l&15, khalf=l>>4; VGPR j<4: K=khalf*8+2j(,+1);
//                  VGPR j>=4: K=16+khalf*8+2(j-4)(,+1)
//   B (32x16 f16), not doc order; A loads match: lane l -> col n=l&15, kgrp=l>>4 covering K=kgrp*16+[0,15];
//                  VGPR j: K=kgrp*16+2j (lo16), +1 (hi16)
//   C/D (16x16 f32): lane l -> col n=l&15; VGPR v -> row v + (l>>4)*8
//
// Wp[kt][nt][lane][j] holds the packed (f16,f16) dword that lands directly in
// B-fragment VGPR j of `lane` for tile (kt,nt): fragment load = 32B LDS read.
__global__ __launch_bounds__(256)
void gemm64_wmma(const float* __restrict__ X, const float* __restrict__ W,
                 float* __restrict__ H, int nrows) {
    __shared__ unsigned int Wp[2][4][32][8];       // 8 KB, B-fragment layout

    // Stage + pack W (f32 -> f16 pair per dword), 8 entries per thread.
    for (int i = threadIdx.x; i < 2048; i += 256) {
        int j    = i & 7;
        int lane = (i >> 3) & 31;
        int nt   = (i >> 8) & 3;
        int kt   = (i >> 10) & 1;
        int n    = lane & 15;
        int kg   = lane >> 4;
        int k    = kt * 32 + kg * 16 + 2 * j;
        int col  = nt * 16 + n;
        union { _Float16 h[2]; unsigned int u; } p;
        p.h[0] = (_Float16)W[k * F_DIM + col];
        p.h[1] = (_Float16)W[(k + 1) * F_DIM + col];
        Wp[kt][nt][lane][j] = p.u;
    }
    __syncthreads();

    const int wave = (blockIdx.x * blockDim.x + threadIdx.x) >> 5;
    const int lane = threadIdx.x & 31;
    const int row0 = wave * 16;
    if (row0 >= nrows) return;                      // whole wave exits together

    const int n  = lane & 15;
    const int kg = lane >> 4;                       // lane half (0/1)

    // ---- B fragments: straight 32B loads from pre-packed LDS ----
    v16h B[2][4];
#pragma unroll
    for (int kt = 0; kt < 2; ++kt)
#pragma unroll
        for (int nt = 0; nt < 4; ++nt)
            B[kt][nt] = *(const v16h*)&Wp[kt][nt][lane][0];

    // ---- A fragments: one 16-row block of X, two 16x32 K-slices ----
    // Per lane: 2 contiguous runs of 16 floats -> 8x float4 loads + cvt_pk.
    const float4* xr4 = (const float4*)(X + (size_t)(row0 + n) * F_DIM);
    v16h A[2];
#pragma unroll
    for (int kt = 0; kt < 2; ++kt) {
        float4 a0 = xr4[kt * 8 + kg * 4 + 0];      // K = kt*32 + kg*16 + 0..3
        float4 a1 = xr4[kt * 8 + kg * 4 + 1];      // K = kt*32 + kg*16 + 4..7
        float4 a2 = xr4[kt * 8 + kg * 4 + 2];      // K = kt*32 + kg*16 + 8..11
        float4 a3 = xr4[kt * 8 + kg * 4 + 3];      // K = kt*32 + kg*16 + 12..15
        A[kt][0]  = (_Float16)a0.x;  A[kt][1]  = (_Float16)a0.y;
        A[kt][2]  = (_Float16)a0.z;  A[kt][3]  = (_Float16)a0.w;
        A[kt][4]  = (_Float16)a1.x;  A[kt][5]  = (_Float16)a1.y;
        A[kt][6]  = (_Float16)a1.z;  A[kt][7]  = (_Float16)a1.w;
        A[kt][8]  = (_Float16)a2.x;  A[kt][9]  = (_Float16)a2.y;
        A[kt][10] = (_Float16)a2.z;  A[kt][11] = (_Float16)a2.w;
        A[kt][12] = (_Float16)a3.x;  A[kt][13] = (_Float16)a3.y;
        A[kt][14] = (_Float16)a3.z;  A[kt][15] = (_Float16)a3.w;
    }

    // ---- 8 WMMAs -> 16x64 f32 tile ----
#pragma unroll
    for (int nt = 0; nt < 4; ++nt) {
        v8f c = {};
        c = __builtin_amdgcn_wmma_f32_16x16x32_f16(false, A[0], false, B[0][nt],
                                                   (short)0, c, false, false);
        c = __builtin_amdgcn_wmma_f32_16x16x32_f16(false, A[1], false, B[1][nt],
                                                   (short)0, c, false, false);
#pragma unroll
        for (int v = 0; v < 8; ++v) {
            int row = row0 + kg * 8 + v;
            H[(size_t)row * F_DIM + nt * 16 + n] = c[v];
        }
    }
}

// ---------------- message passing ----------------

// acc[i,f] = h[i,f] * dinv[i]^2   (self-loop term; also initializes acc)
__global__ void self_loop_init(const float* __restrict__ h,
                               const float* __restrict__ dinv,
                               float* __restrict__ acc, int total) {
    int i = blockIdx.x * 256 + threadIdx.x;
    if (i < total) {
        float di = dinv[i >> 6];
        acc[i] = h[i] * di * di;
    }
}

// One wave per edge; each lane handles 2 features (float2 gather, 2 f32 atomics).
__global__ void edge_scatter(const long long* __restrict__ src,
                             const long long* __restrict__ dst,
                             const float* __restrict__ dinv,
                             const float* __restrict__ h,
                             float* __restrict__ acc, int E) {
    long long tid = (long long)blockIdx.x * 256 + threadIdx.x;
    int e  = (int)(tid >> 5);
    int f2 = (int)(tid & 31);
    if (e >= E) return;
    int s = (int)src[e];
    int d = (int)dst[e];
    float nm = dinv[s] * dinv[d];
    float2 hv = ((const float2*)(h + (size_t)s * F_DIM))[f2];
    float* ap = acc + (size_t)d * F_DIM + 2 * f2;
    atomicAdd(ap,     hv.x * nm);
    atomicAdd(ap + 1, hv.y * nm);
}

__global__ void bias_relu(const float* __restrict__ acc,
                          const float* __restrict__ b,
                          float* __restrict__ out, int total) {
    int i = blockIdx.x * 256 + threadIdx.x;
    if (i < total) out[i] = fmaxf(acc[i] + b[i & (F_DIM - 1)], 0.0f);
}

// ---------------- pooling + readout ----------------

__global__ void pool_zero(float* sums, float* cnt) {
    int i = blockIdx.x * 256 + threadIdx.x;
    if (i < NGRAPH * F_DIM) sums[i] = 0.0f;
    if (i < NGRAPH) cnt[i] = 0.0f;
}

__global__ void pool_accum(const float* __restrict__ x,
                           const long long* __restrict__ batch,
                           float* sums, float* cnt, int total) {
    int i = blockIdx.x * 256 + threadIdx.x;
    if (i >= total) return;
    int node = i >> 6, f = i & (F_DIM - 1);
    int g = (int)batch[node];
    atomicAdd(&sums[g * F_DIM + f], x[i]);
    if (f == 0) atomicAdd(&cnt[g], 1.0f);
}

__global__ void readout(const float* __restrict__ sums, const float* __restrict__ cnt,
                        const float* __restrict__ linW, const float* __restrict__ linb,
                        float* __restrict__ out) {
    __shared__ float red[F_DIM];
    int g = blockIdx.x, t = threadIdx.x;
    float c = fmaxf(cnt[g], 1.0f);
    red[t] = (sums[g * F_DIM + t] / c) * linW[t];
    __syncthreads();
    for (int s = 32; s > 0; s >>= 1) {
        if (t < s) red[t] += red[t + s];
        __syncthreads();
    }
    if (t == 0) out[g] = red[0] + linb[0];
}

// ---------------- host ----------------

extern "C" void kernel_launch(void* const* d_in, const int* in_sizes, int n_in,
                              void* d_out, int out_size, void* d_ws, size_t ws_size,
                              hipStream_t stream) {
    (void)n_in; (void)out_size; (void)ws_size;
    const float*     x     = (const float*)d_in[0];
    const long long* eidx  = (const long long*)d_in[1];   // int64 [2, E] row-major
    const long long* batch = (const long long*)d_in[2];   // int64 [N]
    const float* W[3] = { (const float*)d_in[3], (const float*)d_in[5], (const float*)d_in[7] };
    const float* b[3] = { (const float*)d_in[4], (const float*)d_in[6], (const float*)d_in[8] };
    const float* linW = (const float*)d_in[9];
    const float* linb = (const float*)d_in[10];

    const int N = in_sizes[0] / F_DIM;     // 50000 (multiple of 16)
    const int E = in_sizes[1] / 2;         // 800000
    const long long* src = eidx;
    const long long* dst = eidx + E;

    // workspace carve-out (256B aligned slices)
    char* ws = (char*)d_ws;
    size_t off = 0;
    auto take = [&](size_t bytes) -> void* {
        void* p = ws + off;
        off += (bytes + 255) & ~(size_t)255;
        return p;
    };
    float* deg  = (float*)take((size_t)N * 4);
    float* dinv = (float*)take((size_t)N * 4);
    float* h    = (float*)take((size_t)N * F_DIM * 4);
    float* acc  = (float*)take((size_t)N * F_DIM * 4);
    float* xbuf = (float*)take((size_t)N * F_DIM * 4);
    float* sums = (float*)take((size_t)NGRAPH * F_DIM * 4);
    float* cnt  = (float*)take((size_t)NGRAPH * 4);

    const int TB = 256;
    const int totalNF = N * F_DIM;

    // GCN symmetric normalization with self-loops
    deg_init<<<(N + TB - 1) / TB, TB, 0, stream>>>(deg, N);
    deg_accum<<<(E + TB - 1) / TB, TB, 0, stream>>>(dst, deg, E);
    dinv_kernel<<<(N + TB - 1) / TB, TB, 0, stream>>>(deg, dinv, N);

    const float* cur = x;
    for (int L = 0; L < 3; ++L) {
        int waves  = (N + 15) / 16;                       // 3125
        int blocks = (waves + 7) / 8;                     // 8 waves / block
        gemm64_wmma<<<blocks, 256, 0, stream>>>(cur, W[L], h, N);
        self_loop_init<<<(totalNF + TB - 1) / TB, TB, 0, stream>>>(h, dinv, acc, totalNF);
        long long sthreads = (long long)E * 32;
        edge_scatter<<<(int)((sthreads + TB - 1) / TB), TB, 0, stream>>>(src, dst, dinv, h, acc, E);
        bias_relu<<<(totalNF + TB - 1) / TB, TB, 0, stream>>>(acc, b[L], xbuf, totalNF);
        cur = xbuf;
    }

    pool_zero<<<(NGRAPH * F_DIM + TB - 1) / TB, TB, 0, stream>>>(sums, cnt);
    pool_accum<<<(totalNF + TB - 1) / TB, TB, 0, stream>>>(cur, batch, sums, cnt, totalNF);
    readout<<<NGRAPH, F_DIM, 0, stream>>>(sums, cnt, linW, linb, (float*)d_out);
}
